// PolicyTranslationModel_67345087201728
// MI455X (gfx1250) — matrix operand order repaired and checked by reading, not compiled, not run on hardware
//
#include <hip/hip_runtime.h>
#include <hip/hip_bf16.h>

// MI455X (gfx1250) implementation.
// Strategy: 1 wave32 owns a 16-row batch tile. All row-parallel matmuls are
// v_wmma_f32_16x16x32_f16 (M=16 over batch, K padded to 32). Activations live
// in per-wave LDS arenas as row-major f16 (A-frags = 2x ds_load_b128).
// Weights are pre-swizzled once into B-fragment layout in d_ws by a prep
// kernel (B-frag = 32 lanes x 32B -> 2x global_load_b128 per fragment).
// Scalar glue (softmax/hard-sigmoid/RBF) runs on VALU in lanes<16 regions
// (WMMA never under divergent EXEC).

typedef _Float16 v16h __attribute__((ext_vector_type(16)));
typedef _Float16 v8h  __attribute__((ext_vector_type(8)));
typedef float    v8f  __attribute__((ext_vector_type(8)));

#define DEV __device__ __forceinline__

constexpr int B_    = 8192;
constexpr int SEQ   = 16;
constexpr int FEAT  = 5;
constexpr int NREG  = 6;
constexpr int EMB   = 50;
constexpr int U     = 32;
constexpr int OD    = 7;
constexpr int NBF   = 11;
constexpr float SCL = 0.012f;

// ---- ws (scratch) B-fragment layout: 1 frag = 512 halves = 1KB ----
constexpr int FR      = 512;
constexpr int F_LNGK  = 0;        // KB=2 NB=6 -> 12 frags
constexpr int F_LNGRK = 12 * FR;  // KB=1 NB=6 -> 6
constexpr int F_ATTW2 = 18 * FR;  // KB=1 NB=4 -> 4
constexpr int F_PTG   = 22 * FR;  // KB=2 NB=3 -> 6
constexpr int F_DT1   = 28 * FR;  // KB=2 NB=4 -> 8
constexpr int F_ROBK  = 36 * FR;  // KB=1 NB=6 -> 6
constexpr int F_ROBRK = 42 * FR;  // KB=1 NB=6 -> 6
constexpr int F_W1    = 48 * FR;  // KB=3 NB=5 -> 15
constexpr int F_W2    = 63 * FR;  // 15
constexpr int F_W3    = 78 * FR;  // 15
constexpr int F_PH1   = 93 * FR;  // KB=3 NB=1 -> 3   (total 96 frags = 96KB)

// ---- per-wave LDS arena (bytes) ----
constexpr int A_HL   = 0;      // lang GRU h: 16x32 f16
constexpr int A_HR   = 1024;   // robot GRU h: 16x32 f16
constexpr int A_LW2  = 2048;   // lang@attn_w2: 16x64 f32
constexpr int A_LOG  = 6144;   // attn logits: 16x8 f32
constexpr int A_X1   = 6656;   // cfeat|lang:  16x64 f16
constexpr int A_PT   = 8704;   // ptg out:     16x64 f16
constexpr int A_DT1  = 10752;  // dt1 out:     16x64 f32
constexpr int A_X96  = 14848;  // x concat:    16x96 f16
constexpr int A_WB   = 17920;  // w chain:     16x96 f16
constexpr int A_W3F  = 20992;  // w3 out:      16x80 f32
constexpr int A_PH1F = 26112;  // ph1 out:     16x16 f32
constexpr int ARENA  = 27136;

DEV float sigm(float x)  { return 1.0f / (1.0f + __expf(-x)); }
DEV float hsig(float x)  { return fminf(fmaxf(0.2f * x + 0.5f, 0.0f), 1.0f); }

DEV v8f vzero() {
  v8f z;
#pragma unroll
  for (int i = 0; i < 8; ++i) z[i] = 0.0f;
  return z;
}

DEV v8f wmma_f16(v16h a, v16h b, v8f c) {
  // emits v_wmma_f32_16x16x32_f16
  return __builtin_amdgcn_wmma_f32_16x16x32_f16(false, a, false, b, (short)0, c,
                                                false, false);
}

// B-fragment from pre-swizzled global scratch: lane gets 32 contiguous bytes.
DEV v16h load_bfrag(const _Float16* ws, int frag, int lane) {
  const v8h* p = (const v8h*)(ws + (long)frag * FR + lane * 16);
  union { v16h v; v8h h[2]; } u;
  u.h[0] = p[0];
  u.h[1] = p[1];
  return u.v;
}

// A-fragment (16x32 f16, M=row) from row-major f16 LDS [16][ldh], K block kb.
// ISA layout: element j <-> k = (j<8?0:16) + 8*half + (j&7), M = lane&15.
DEV v16h load_afrag_lds(const _Float16* act, int ldh, int kb, int lane) {
  const int row = lane & 15, hf = lane >> 4;
  const _Float16* b = act + row * ldh + kb * 32 + 8 * hf;
  union { v16h v; v8h h[2]; } u;
  u.h[0] = *(const v8h*)(b);
  u.h[1] = *(const v8h*)(b + 16);
  return u.v;
}

// A-fragment gathered straight from a per-row f32 global vector (embeddings /
// robot input), zero-padded past kreal.
DEV v16h load_afrag_gmem(const float* rowp, int kb, int kreal, int lane) {
  const int hf = lane >> 4;
  v16h v;
#pragma unroll
  for (int j = 0; j < 16; ++j) {
    int k = kb * 32 + ((j < 8) ? 0 : 16) + 8 * hf + (j & 7);
    v[j] = (k < kreal) ? (_Float16)rowp[k] : (_Float16)0.0f;
  }
  return v;
}

// One keras GRU step (reset_after=True, gates [z,r,h]) on a 16-row tile.
// x-projection accumulators come from ax[] (KBX K-blocks), h-projection from
// ah. Weight frags indexed [kb*6+nb] (x) / [nb] (h). h state in hlds (f16).
template <int KBX>
DEV void gru_core(const v16h* ax, v16h ah, const _Float16* wsX,
                  const _Float16* wsH, const float* b2x96, _Float16* hlds,
                  int lane, float* erob_out, int b0) {
  const int l15 = lane & 15, hf = lane >> 4;
  float rsv[16], hsv[16];

  // r gates: gate cols 32..63 -> nb = 2,3
#pragma unroll
  for (int i = 0; i < 2; ++i) {
    const int nb = 2 + i;
    v8f xacc = vzero(), hacc = vzero();
#pragma unroll
    for (int kb = 0; kb < KBX; ++kb)
      xacc = wmma_f16(ax[kb], load_bfrag(wsX, kb * 6 + nb, lane), xacc);
    hacc = wmma_f16(ah, load_bfrag(wsH, nb, lane), hacc);
    const int n = nb * 16 + l15;
    const float bx = b2x96[n], bh = b2x96[96 + n];
#pragma unroll
    for (int r = 0; r < 8; ++r)
      rsv[i * 8 + r] = sigm(xacc[r] + bx + hacc[r] + bh);
  }

  // hh candidates: gate cols 64..95 -> nb = 4,5 ; hh = tanh(xz + r*hz)
#pragma unroll
  for (int i = 0; i < 2; ++i) {
    const int nb = 4 + i;
    v8f xacc = vzero(), hacc = vzero();
#pragma unroll
    for (int kb = 0; kb < KBX; ++kb)
      xacc = wmma_f16(ax[kb], load_bfrag(wsX, kb * 6 + nb, lane), xacc);
    hacc = wmma_f16(ah, load_bfrag(wsH, nb, lane), hacc);
    const int n = nb * 16 + l15;
    const float bx = b2x96[n], bh = b2x96[96 + n];
#pragma unroll
    for (int r = 0; r < 8; ++r)
      hsv[i * 8 + r] = tanhf(xacc[r] + bx + rsv[i * 8 + r] * (hacc[r] + bh));
  }

  // z gates + state update: gate cols 0..31 -> nb = 0,1
#pragma unroll
  for (int i = 0; i < 2; ++i) {
    const int nb = i;
    v8f xacc = vzero(), hacc = vzero();
#pragma unroll
    for (int kb = 0; kb < KBX; ++kb)
      xacc = wmma_f16(ax[kb], load_bfrag(wsX, kb * 6 + nb, lane), xacc);
    hacc = wmma_f16(ah, load_bfrag(wsH, nb, lane), hacc);
    const int n = nb * 16 + l15;
    const float bx = b2x96[n], bh = b2x96[96 + n];
#pragma unroll
    for (int r = 0; r < 8; ++r) {
      const float z = sigm(xacc[r] + bx + hacc[r] + bh);
      const int row = 8 * hf + r;
      const float hold = (float)hlds[row * U + n];
      const float hn = z * hold + (1.0f - z) * hsv[i * 8 + r];
      hlds[row * U + n] = (_Float16)hn;
      if (erob_out) erob_out[(long)(b0 + row) * U + n] = hn;
    }
  }
}

// Generic 16-row GEMM tile: out = act(16xK f16 LDS) @ W(+bias), optional relu,
// f16 or f32 sink. Cols >= Nreal are written as exact zeros (padding).
template <int KB, int NB, bool RELU, bool OUT16>
DEV void gemm_tile(const _Float16* act, int ldh, const _Float16* frags,
                   const float* bias, int Nreal, _Float16* o16, float* o32,
                   int outld, int lane) {
  v16h a[KB];
#pragma unroll
  for (int kb = 0; kb < KB; ++kb) a[kb] = load_afrag_lds(act, ldh, kb, lane);
  const int l15 = lane & 15, hf = lane >> 4;
#pragma unroll
  for (int nb = 0; nb < NB; ++nb) {
    v8f acc = vzero();
#pragma unroll
    for (int kb = 0; kb < KB; ++kb)
      acc = wmma_f16(a[kb], load_bfrag(frags, kb * NB + nb, lane), acc);
    const int n = nb * 16 + l15;
    const float bv = (n < Nreal) ? bias[n] : 0.0f;
#pragma unroll
    for (int r = 0; r < 8; ++r) {
      const int row = 8 * hf + r;
      float v = (n < Nreal) ? (acc[r] + bv) : 0.0f;
      if (RELU) v = fmaxf(v, 0.0f);
      if (OUT16) o16[row * outld + n] = (_Float16)v;
      else       o32[row * outld + n] = v;
    }
  }
}

// Prep: convert f32 weight [K][N] into swizzled f16 B-fragments.
// B-frag layout: lane = (half,n), element j <-> k = kb*32 + (j<8?0:16) + 8*half + (j&7).
__global__ void prep_frag(const float* __restrict__ src, int K, int N, int NB,
                          _Float16* __restrict__ dst) {
  const int f = blockIdx.x, lane = threadIdx.x;
  const int kb = f / NB, nb = f % NB;
  const int hf = lane >> 4, n = nb * 16 + (lane & 15);
  _Float16* d = dst + (long)f * FR + lane * 16;
#pragma unroll
  for (int j = 0; j < 16; ++j) {
    const int k = kb * 32 + ((j < 8) ? 0 : 16) + 8 * hf + (j & 7);
    d[j] = (k < K && n < N) ? (_Float16)src[(long)k * N + n] : (_Float16)0.0f;
  }
}

__global__ __launch_bounds__(64) void policy_main(
    const int* __restrict__ language, const float* __restrict__ features,
    const float* __restrict__ robot, const float* __restrict__ st_robot_last,
    const float* __restrict__ st_gru, const float* __restrict__ emb_table,
    const float* __restrict__ lng_b, const float* __restrict__ attn_w1,
    const float* __restrict__ attn_b1, const float* __restrict__ attn_b2,
    const float* __restrict__ attn_wt, const float* __restrict__ attn_bt,
    const float* __restrict__ ptg_b, const float* __restrict__ dt1_b,
    const float* __restrict__ dt2_w, const float* __restrict__ dt2_b,
    const float* __restrict__ rob_b, const float* __restrict__ w1_b,
    const float* __restrict__ w2_b, const float* __restrict__ w3_b,
    const float* __restrict__ ph1_b, const float* __restrict__ ph2_w,
    const float* __restrict__ ph2_b, const _Float16* __restrict__ ws,
    float* __restrict__ out) {
  extern __shared__ __align__(16) char smem[];
  const int lane = threadIdx.x & 31;
  const int wave = threadIdx.x >> 5;
  const int tile = blockIdx.x * 2 + wave;
  const int b0 = tile * 16;
  char* arena = smem + wave * ARENA;

  _Float16* hL   = (_Float16*)(arena + A_HL);
  _Float16* hR   = (_Float16*)(arena + A_HR);
  float*    lw2  = (float*)(arena + A_LW2);
  float*    lg   = (float*)(arena + A_LOG);
  _Float16* x1   = (_Float16*)(arena + A_X1);
  _Float16* pt   = (_Float16*)(arena + A_PT);
  float*    dt1f = (float*)(arena + A_DT1);
  _Float16* x96  = (_Float16*)(arena + A_X96);
  _Float16* wb   = (_Float16*)(arena + A_WB);
  float*    w3f  = (float*)(arena + A_W3F);
  float*    ph1f = (float*)(arena + A_PH1F);

  // Zero the whole arena (GRU h0 = 0, and all K/N pads become exact zeros).
  for (int i = lane; i < ARENA / 4; i += 32) ((unsigned*)arena)[i] = 0u;
  asm volatile("s_wait_dscnt 0" : : : "memory");

  const int row = lane & 15;

  // ---------------- language GRU over SEQ steps ----------------
  for (int t = 0; t < SEQ; ++t) {
    const int idx = language[(long)(b0 + row) * SEQ + t];
    const float* er = emb_table + (long)idx * EMB;
    if (t + 1 < SEQ) {
      const int idx2 = language[(long)(b0 + row) * SEQ + t + 1];
      __builtin_prefetch(emb_table + (long)idx2 * EMB, 0, 1);  // global_prefetch_b8
    }
    v16h ax[2];
    ax[0] = load_afrag_gmem(er, 0, EMB, lane);
    ax[1] = load_afrag_gmem(er, 1, EMB, lane);
    const v16h ah = load_afrag_lds(hL, U, 0, lane);
    gru_core<2>(ax, ah, ws + F_LNGK, ws + F_LNGRK, lng_b, hL, lane, nullptr, b0);
  }

  // ---------------- top-down attention ----------------
  // lw2 = lang @ attn_w2 + b2   (WMMA)
  gemm_tile<1, 4, false, false>(hL, U, ws + F_ATTW2, attn_b2, 64, nullptr, lw2,
                                64, lane);
  asm volatile("s_wait_dscnt 0" : : : "memory");
  // per-(row,region) logit = bt + wt . tanh(f@W1 + b1 + lw2[row])  (VALU)
  for (int p = lane; p < 16 * NREG; p += 32) {
    const int rw = p / NREG, rg = p % NREG;
    const float* fp = features + ((long)(b0 + rw) * NREG + rg) * FEAT;
    float acc = attn_bt[0];
    for (int o = 0; o < 64; ++o) {
      float s = attn_b1[o] + lw2[rw * 64 + o];
#pragma unroll
      for (int k = 0; k < FEAT; ++k) s += fp[k] * attn_w1[k * 64 + o];
      acc += attn_wt[o] * tanhf(s);
    }
    lg[rw * 8 + rg] = acc;
  }
  asm volatile("s_wait_dscnt 0" : : : "memory");
  if (lane < 16) {
    const int m = lane;
    float mx = -1e30f;
    for (int r = 0; r < NREG; ++r) mx = fmaxf(mx, lg[m * 8 + r]);
    float e[NREG], s = 0.0f;
    for (int r = 0; r < NREG; ++r) { e[r] = __expf(lg[m * 8 + r] - mx); s += e[r]; }
    float cf[FEAT] = {0.f, 0.f, 0.f, 0.f, 0.f};
    for (int r = 0; r < NREG; ++r) {
      const float a = e[r] / s;
      out[85 * B_ + (long)(b0 + m) * NREG + r] = a;  // atn
      const float* fp = features + ((long)(b0 + m) * NREG + r) * FEAT;
#pragma unroll
      for (int k = 0; k < FEAT; ++k) cf[k] += a * fp[k];
    }
    for (int k = 0; k < FEAT; ++k) x1[m * 64 + k] = (_Float16)cf[k];
    for (int c = 0; c < U; ++c) x1[m * 64 + FEAT + c] = hL[m * U + c];
  }

  // ---------------- ptg -> dt1 (WMMA) ----------------
  gemm_tile<2, 3, true, true>(x1, 64, ws + F_PTG, ptg_b, 42, pt, nullptr, 64, lane);
  gemm_tile<2, 4, true, false>(pt, 64, ws + F_DT1, dt1_b, 64, nullptr, dt1f, 64, lane);

  // ---------------- robot GRU (single step, WMMA) ----------------
  for (int i = lane; i < 16 * U; i += 32) {
    const int rr = i / U, cc = i % U;
    hR[i] = (_Float16)st_gru[(long)(b0 + rr) * U + cc];
  }
  {
    v16h axr[1];
    axr[0] = load_afrag_gmem(robot + (long)(b0 + row) * OD, 0, OD, lane);
    const v16h ahr = load_afrag_lds(hR, U, 0, lane);
    gru_core<1>(axr, ahr, ws + F_ROBK, ws + F_ROBRK, rob_b, hR, lane,
                out + 92 * B_, b0);  // e_robot output
  }

  // ---------------- x = [cfeat | lang | e_robot | st_robot_last] ----------------
  asm volatile("s_wait_dscnt 0" : : : "memory");
  if (lane < 16) {
    const int m = lane;
    for (int k = 0; k < FEAT + U; ++k) x96[m * 96 + k] = x1[m * 64 + k];
    for (int c = 0; c < U; ++c) x96[m * 96 + FEAT + U + c] = hR[m * U + c];
    for (int k = 0; k < OD; ++k)
      x96[m * 96 + FEAT + 2 * U + k] =
          (_Float16)st_robot_last[(long)(b0 + m) * OD + k];
  }

  // ---------------- w chain + ph1 (WMMA) ----------------
  gemm_tile<3, 5, true,  true >(x96, 96, ws + F_W1, w1_b, 77, wb, nullptr, 96, lane);
  gemm_tile<3, 5, true,  true >(wb,  96, ws + F_W2, w2_b, 77, wb, nullptr, 96, lane);
  gemm_tile<3, 5, false, false>(wb,  96, ws + F_W3, w3_b, 77, nullptr, w3f, 80, lane);
  gemm_tile<3, 1, true,  false>(x96, 96, ws + F_PH1, ph1_b, 16, nullptr, ph1f, 16, lane);
  asm volatile("s_wait_dscnt 0" : : : "memory");

  // ---------------- per-row scalar tail ----------------
  if (lane < 16) {
    const int m = lane;
    const int b = b0 + m;
    float s = dt2_b[0];
    for (int o = 0; o < 64; ++o) s += dt1f[m * 64 + o] * dt2_w[o];
    const float dmp = hsig(s) + 0.1f;
    out[91 * B_ + b] = dmp;  // dmp_dt
    const float dt = 1.0f / (500.0f * dmp);

    float s2 = ph2_b[0];
    for (int o = 0; o < 16; ++o) s2 += ph1f[m * 16 + o] * ph2_w[o];
    const float phase = hsig(s2) + dt;
    out[7 * B_ + b] = phase;  // phase

    float psi[NBF], ps = 0.0f;
#pragma unroll
    for (int k = 0; k < NBF; ++k) {
      const float d = phase - 0.1f * (float)k;
      psi[k] = __expf(-(d * d) / (2.0f * SCL));
      ps += psi[k];
    }
    const float inv = 1.0f / ps;
    for (int d = 0; d < OD; ++d) {
      float av = 0.0f;
#pragma unroll
      for (int k = 0; k < NBF; ++k) {
        const float wv = w3f[m * 80 + d * NBF + k];
        out[8 * B_ + (long)b * (OD * NBF) + d * NBF + k] = wv;  // weights
        av += wv * psi[k] * inv;
      }
      out[(long)b * OD + d] = av;  // action
    }
  }
}

extern "C" void kernel_launch(void* const* d_in, const int* in_sizes, int n_in,
                              void* d_out, int out_size, void* d_ws,
                              size_t ws_size, hipStream_t stream) {
  (void)in_sizes; (void)n_in; (void)out_size; (void)ws_size;
  _Float16* ws = (_Float16*)d_ws;
  float* out = (float*)d_out;

  auto P = [&](int idx, int K, int N, int KB, int NB, int fbase) {
    prep_frag<<<KB * NB, 32, 0, stream>>>((const float*)d_in[idx], K, N, NB,
                                          ws + fbase);
  };
  P(6,  50, 96, 2, 6, F_LNGK);
  P(7,  32, 96, 1, 6, F_LNGRK);
  P(11, 32, 64, 1, 4, F_ATTW2);
  P(15, 37, 42, 2, 3, F_PTG);
  P(17, 42, 64, 2, 4, F_DT1);
  P(21,  7, 96, 1, 6, F_ROBK);
  P(22, 32, 96, 1, 6, F_ROBRK);
  P(24, 76, 77, 3, 5, F_W1);
  P(26, 77, 77, 3, 5, F_W2);
  P(28, 77, 77, 3, 5, F_W3);
  P(30, 76, 16, 3, 1, F_PH1);

  const size_t smem = 2 * (size_t)ARENA;  // 2 waves/block
  policy_main<<<B_ / 16 / 2, 64, smem, stream>>>(
      (const int*)d_in[0], (const float*)d_in[1], (const float*)d_in[2],
      (const float*)d_in[3], (const float*)d_in[4], (const float*)d_in[5],
      (const float*)d_in[8],  /* lng_b   */
      (const float*)d_in[9],  /* attn_w1 */
      (const float*)d_in[10], /* attn_b1 */
      (const float*)d_in[12], /* attn_b2 */
      (const float*)d_in[13], /* attn_wt */
      (const float*)d_in[14], /* attn_bt */
      (const float*)d_in[16], /* ptg_b   */
      (const float*)d_in[18], /* dt1_b   */
      (const float*)d_in[19], /* dt2_w   */
      (const float*)d_in[20], /* dt2_b   */
      (const float*)d_in[23], /* rob_b   */
      (const float*)d_in[25], /* w1_b    */
      (const float*)d_in[27], /* w2_b    */
      (const float*)d_in[29], /* w3_b    */
      (const float*)d_in[31], /* ph1_b   */
      (const float*)d_in[32], /* ph2_w   */
      (const float*)d_in[33], /* ph2_b   */
      ws, out);
}
